// SelectiveScanSSM_58188216926725
// MI455X (gfx1250) — matrix operand order repaired and checked by reading, not compiled, not run on hardware
//
#include <hip/hip_runtime.h>
#include <hip/hip_bf16.h>

// ---------------------------------------------------------------------------
// Selective-scan SSM (Mamba block) for gfx1250.
//  - Big GEMMs: v_wmma_f32_16x16x32_bf16, f32 accumulate.
//  - Tile staging: Tensor Data Mover (tensor_load_to_lds), double buffered,
//    TDM applies the LDS row padding (pad_interval=16 DW, pad_amount=4 DW).
//  - Weights pre-transposed to [N][K] bf16 so both A and B fragments are
//    contiguous 16-lane-element LDS reads (ds_load_b128 pairs).
//  - Scan: register-resident h[16] per channel, B/C tiles staged in LDS.
// ---------------------------------------------------------------------------

typedef __attribute__((ext_vector_type(16))) __bf16 v16bf;
typedef __attribute__((ext_vector_type(8)))  float  v8f;
typedef __attribute__((ext_vector_type(4)))  unsigned int u32x4;
typedef __attribute__((ext_vector_type(8)))  int          i32x8;
typedef __attribute__((ext_vector_type(4)))  int          i32x4;

#define D_MODEL   1024
#define STATE_DIM 16
#define DT_RANK   64
#define D_INNER   2048
#define BATCH     4
#define SEQ_LEN   2048
#define NT        (BATCH * SEQ_LEN)   // 8192 rows

enum { EPI_SPLIT = 0, EPI_SOFTPLUS = 1, EPI_BIAS = 2 };

// ---------------------------------------------------------------------------
// TDM 2D tile load: global (row-major, row_stride elems of bf16) -> LDS,
// with LDS padding of 4 DWORDs every 16 DWORDs (i.e. 32 bf16 row -> 40 bf16).
// D# per CDNA5 ISA ch.8. Issued by whichever wave executes it (EXEC ignored).
// This toolchain exposes the 6-arg builtin:
//   (u32x4 g0, i32x8 g1, i32x4 g2, i32x4 g3, i32x8 extra, i32 cpol)
// ---------------------------------------------------------------------------
__device__ __forceinline__ void tdm_load_2d_bf16(unsigned lds_addr,
                                                 const void* gptr,
                                                 unsigned tile_w,   // elems
                                                 unsigned tile_h,   // rows
                                                 unsigned row_stride) { // elems
  const unsigned long long ga = (unsigned long long)(uintptr_t)gptr;
  u32x4 g0;
  g0[0] = 1u;                                   // count=1 (valid), user mode
  g0[1] = lds_addr;                             // lds_addr [63:32]
  g0[2] = (unsigned)(ga & 0xFFFFFFFFu);         // global_addr[31:0]
  g0[3] = (unsigned)((ga >> 32) & 0x1FFFFFFu)   // global_addr[56:32]
          | (2u << 30);                         // type=2 ("image")
  i32x8 g1;
  // data_size=1 (2B) | pad_enable | pad_interval=3 (16 DW) | pad_amount=3 (4 DW)
  g1[0] = (int)((1u << 16) | (1u << 20) | (3u << 22) | (3u << 25));
  const unsigned tw = row_stride;               // tensor_dim0: full row length
  const unsigned th = 0x100000u;                // tensor_dim1: large (no clip)
  g1[1] = (int)((tw & 0xFFFFu) << 16);          // atomic_addr=0 | dim0 lo
  g1[2] = (int)(((tw >> 16) & 0xFFFFu) | ((th & 0xFFFFu) << 16));
  g1[3] = (int)(((th >> 16) & 0xFFFFu) | ((tile_w & 0xFFFFu) << 16));
  g1[4] = (int)(tile_h & 0xFFFFu);              // tile_dim1 | tile_dim2=0
  g1[5] = (int)row_stride;                      // tensor_dim0_stride[31:0]
  g1[6] = 0;                                    // stride hi | dim1_stride lo
  g1[7] = 0;
  const i32x4 z4 = {0, 0, 0, 0};
  const i32x8 z8 = {0, 0, 0, 0, 0, 0, 0, 0};
  __builtin_amdgcn_tensor_load_to_lds(g0, g1, z4, z4, z8, 0);
}

__device__ __forceinline__ unsigned lds_addr_of(const void* p) {
  // flat LDS pointers carry the aperture in [63:32]; low 32 bits = LDS offset
  return (unsigned)(uintptr_t)p;
}

// ---------------------------------------------------------------------------
// f32 -> bf16 elementwise conversion
// ---------------------------------------------------------------------------
__global__ void cvt_bf16_kernel(const float* __restrict__ src,
                                __bf16* __restrict__ dst, int n) {
  int i = blockIdx.x * 256 + threadIdx.x;
  if (i < n) dst[i] = (__bf16)src[i];
}

// ---------------------------------------------------------------------------
// f32 [K][N] -> bf16 [N][K] transpose (LDS-tiled 32x32, 256 threads = 32x8)
// ---------------------------------------------------------------------------
__global__ __launch_bounds__(256)
void cvt_transpose_kernel(const float* __restrict__ src, __bf16* __restrict__ dst,
                          int K, int N) {
  __shared__ float t[32][33];
  const int kb = blockIdx.x * 32, nb = blockIdx.y * 32;
  const int tx = threadIdx.x & 31, ty = threadIdx.x >> 5;
#pragma unroll
  for (int j = 0; j < 32; j += 8)
    t[ty + j][tx] = src[(size_t)(kb + ty + j) * N + nb + tx];
  __syncthreads();
#pragma unroll
  for (int j = 0; j < 32; j += 8)
    dst[(size_t)(nb + ty + j) * K + kb + tx] = (__bf16)t[tx][ty + j];
}

// ---------------------------------------------------------------------------
// WMMA GEMM: C[M,N] = A[M,K](bf16) * BT[N,K](bf16)^T + bias, fused epilogue.
// Block tile 64x128, 8 waves (2x4), per-wave 32x32 (4 wmma / k-step).
// A and B tiles staged by TDM, double buffered.
// ---------------------------------------------------------------------------
template <int EPI>
__global__ __launch_bounds__(256)
void wmma_gemm_kernel(const __bf16* __restrict__ A,
                      const __bf16* __restrict__ BT,
                      const float* __restrict__ bias,
                      float* __restrict__ out0,
                      float* __restrict__ out1,
                      int M, int N, int K, int lda, int ldb) {
  // padded rows: 32 bf16 data + 8 bf16 pad (TDM pad config matches)
  __shared__ __align__(16) __bf16 sA[2][64][40];
  __shared__ __align__(16) __bf16 sB[2][128][40];

  const int tid  = threadIdx.x;
  const int wave = tid >> 5;
  const int lane = tid & 31;
  const int l    = lane & 15;
  const int hi   = lane >> 4;
  const int wm   = wave & 1;    // 2 waves along M
  const int wn   = wave >> 1;   // 4 waves along N
  const int rowBase = blockIdx.x * 64;
  const int colBase = blockIdx.y * 128;

  v8f acc[2][2];
  const v8f vzero = {0.f, 0.f, 0.f, 0.f, 0.f, 0.f, 0.f, 0.f};
  acc[0][0] = vzero; acc[0][1] = vzero; acc[1][0] = vzero; acc[1][1] = vzero;

  const __bf16* Abase = A  + (size_t)rowBase * lda;
  const __bf16* Bbase = BT + (size_t)colBase * ldb;

  // prologue: stage k0=0 into buffer 0 (wave 0 only; TDM ignores EXEC)
  if (tid < 32) {
    tdm_load_2d_bf16(lds_addr_of(&sA[0][0][0]), Abase, 32, 64, (unsigned)lda);
    tdm_load_2d_bf16(lds_addr_of(&sB[0][0][0]), Bbase, 32, 128, (unsigned)ldb);
  }

  int buf = 0;
  for (int k0 = 0; k0 < K; k0 += 32) {
    __syncthreads();               // all reads of buf^1 (prev iter) complete
    if (tid < 32) {
      if (k0 + 32 < K) {           // prefetch next tile pair into other buffer
        tdm_load_2d_bf16(lds_addr_of(&sA[buf ^ 1][0][0]), Abase + (k0 + 32),
                         32, 64, (unsigned)lda);
        tdm_load_2d_bf16(lds_addr_of(&sB[buf ^ 1][0][0]), Bbase + (k0 + 32),
                         32, 128, (unsigned)ldb);
        __builtin_amdgcn_s_wait_tensorcnt(2);  // current pair done, next in flight
      } else {
        __builtin_amdgcn_s_wait_tensorcnt(0);
      }
    }
    __syncthreads();               // publish current buffer to all waves

    // A fragment (16x32): lane row M=l; lanes 0-15 K 0..7 & 16..23,
    // lanes 16-31 K 8..15 & 24..31 -> two contiguous 8-element runs.
    v16bf af[2];
#pragma unroll
    for (int mi = 0; mi < 2; ++mi) {
      const int r = wm * 32 + mi * 16 + l;
#pragma unroll
      for (int e = 0; e < 8; ++e) {
        af[mi][e]     = sA[buf][r][hi * 8 + e];
        af[mi][e + 8] = sA[buf][r][16 + hi * 8 + e];
      }
    }
    // B fragment (32x16): lane col N=l; lanes 0-15 K 0..7 & 16..23,
    // lanes 16-31 K 8..15 & 24..31 -> two contiguous 8-element runs.
    v16bf bfr[2];
#pragma unroll
    for (int ni = 0; ni < 2; ++ni) {
      const int c = wn * 32 + ni * 16 + l;
#pragma unroll
      for (int e = 0; e < 8; ++e) {
        bfr[ni][e]     = sB[buf][c][hi * 8 + e];
        bfr[ni][e + 8] = sB[buf][c][16 + hi * 8 + e];
      }
    }

#pragma unroll
    for (int mi = 0; mi < 2; ++mi)
#pragma unroll
      for (int ni = 0; ni < 2; ++ni)
        acc[mi][ni] = __builtin_amdgcn_wmma_f32_16x16x32_bf16(
            false, af[mi], false, bfr[ni], (short)0, acc[mi][ni], false, false);
    buf ^= 1;
  }

  // epilogue: C/D layout -> VGPR r holds row M = hi*8 + r, col N = l
#pragma unroll
  for (int mi = 0; mi < 2; ++mi) {
#pragma unroll
    for (int ni = 0; ni < 2; ++ni) {
      const int col = colBase + wn * 32 + ni * 16 + l;
      const float bv = bias[col];
#pragma unroll
      for (int r = 0; r < 8; ++r) {
        const int row = rowBase + wm * 32 + mi * 16 + hi * 8 + r;
        float v = acc[mi][ni][r] + bv;
        if (EPI == EPI_SPLIT) {
          const int half = N >> 1;   // 2048
          if (col < half) out0[(size_t)row * half + col] = v;          // x_in
          else            out1[(size_t)row * half + (col - half)] = v; // z
        } else if (EPI == EPI_SOFTPLUS) {
          float s = (v > 20.f) ? v : log1pf(__expf(v));
          out0[(size_t)row * N + col] = s;                              // dt
        } else {
          out0[(size_t)row * N + col] = v;                              // final
        }
      }
    }
  }
}

// ---------------------------------------------------------------------------
// x_dbl = x_in @ W_x  (N=96): one block per row; row cached in LDS.
// Emits dt_r (bf16), B (f32), C (f32) directly.
// ---------------------------------------------------------------------------
__global__ __launch_bounds__(128)
void xdbl_kernel(const float* __restrict__ x_in, const float* __restrict__ Wx,
                 __bf16* __restrict__ dtr_b, float* __restrict__ Bm,
                 float* __restrict__ Cm) {
  __shared__ __align__(16) float srow[D_INNER];
  const int row = blockIdx.x;
  const int tid = threadIdx.x;
  const size_t base = (size_t)row * D_INNER;
#pragma unroll
  for (int j = 0; j < 4; ++j)
    ((float4*)srow)[tid + j * 128] = ((const float4*)(x_in + base))[tid + j * 128];
  __syncthreads();
  if (tid < 96) {
    float acc = 0.f;
    for (int k = 0; k < D_INNER; ++k) acc = fmaf(srow[k], Wx[k * 96 + tid], acc);
    if (tid < DT_RANK)
      dtr_b[(size_t)row * DT_RANK + tid] = (__bf16)acc;
    else if (tid < DT_RANK + STATE_DIM)
      Bm[(size_t)row * STATE_DIM + (tid - DT_RANK)] = acc;
    else
      Cm[(size_t)row * STATE_DIM + (tid - DT_RANK - STATE_DIM)] = acc;
  }
}

// ---------------------------------------------------------------------------
// Selective scan: one thread per (batch, d) channel; h[16] in registers;
// B/C timestep tiles staged in LDS (shared by all channels of the batch).
// ---------------------------------------------------------------------------
__global__ __launch_bounds__(256)
void scan_kernel(const float* __restrict__ u, const float* __restrict__ dt,
                 const float* __restrict__ Bm, const float* __restrict__ Cm,
                 const float* __restrict__ A_log, const float* __restrict__ Dv,
                 float* __restrict__ y) {
  __shared__ __align__(16) float sB[64 * STATE_DIM];
  __shared__ __align__(16) float sC[64 * STATE_DIM];
  const int tid = threadIdx.x;
  const int b   = blockIdx.x >> 3;                 // 8 blocks per batch
  const int d   = ((blockIdx.x & 7) << 8) + tid;   // channel
  float Arow[STATE_DIM], h[STATE_DIM];
#pragma unroll
  for (int n = 0; n < STATE_DIM; ++n) {
    Arow[n] = -__expf(A_log[d * STATE_DIM + n]);
    h[n] = 0.f;
  }
  const float Dd = Dv[d];
  const size_t rbase = (size_t)b * SEQ_LEN;

  for (int tc = 0; tc < SEQ_LEN; tc += 64) {
    ((float4*)sB)[tid] = ((const float4*)(Bm + (rbase + tc) * STATE_DIM))[tid];
    ((float4*)sC)[tid] = ((const float4*)(Cm + (rbase + tc) * STATE_DIM))[tid];
    __syncthreads();
#pragma unroll 4
    for (int tt = 0; tt < 64; ++tt) {
      const size_t idx = (rbase + tc + tt) * D_INNER + d;
      const float ut  = u[idx];
      const float dtt = dt[idx];
      const float dtu = dtt * ut;
      float yv = 0.f;
#pragma unroll
      for (int n = 0; n < STATE_DIM; ++n) {
        const float hb = __expf(Arow[n] * dtt) * h[n] + dtu * sB[tt * STATE_DIM + n];
        h[n] = hb;
        yv += hb * sC[tt * STATE_DIM + n];
      }
      y[idx] = yv + Dd * ut;
    }
    __syncthreads();
  }
}

// ---------------------------------------------------------------------------
// LayerNorm over D_INNER + SiLU(z) gate; emits bf16 for the final WMMA GEMM.
// ---------------------------------------------------------------------------
__global__ __launch_bounds__(256)
void ln_gate_kernel(const float* __restrict__ y, const float* __restrict__ z,
                    const float* __restrict__ g, const float* __restrict__ bn,
                    __bf16* __restrict__ out) {
  __shared__ float red[256];
  const int row = blockIdx.x;
  const int tid = threadIdx.x;
  const size_t base = (size_t)row * D_INNER;
  float v[8];
  float s = 0.f;
#pragma unroll
  for (int j = 0; j < 8; ++j) {
    v[j] = y[base + tid + j * 256];
    s += v[j];
  }
  red[tid] = s;
  __syncthreads();
  for (int off = 128; off > 0; off >>= 1) {
    if (tid < off) red[tid] += red[tid + off];
    __syncthreads();
  }
  const float mu = red[0] * (1.f / D_INNER);
  __syncthreads();
  float s2 = 0.f;
#pragma unroll
  for (int j = 0; j < 8; ++j) {
    const float dv = v[j] - mu;
    s2 += dv * dv;
  }
  red[tid] = s2;
  __syncthreads();
  for (int off = 128; off > 0; off >>= 1) {
    if (tid < off) red[tid] += red[tid + off];
    __syncthreads();
  }
  const float rstd = rsqrtf(red[0] * (1.f / D_INNER) + 1e-5f);
#pragma unroll
  for (int j = 0; j < 8; ++j) {
    const int c = tid + j * 256;
    const float zn   = z[base + c];
    const float gate = zn / (1.f + __expf(-zn));
    const float o    = (v[j] - mu) * rstd * g[c] + bn[c];
    out[base + c] = (__bf16)(o * gate);
  }
}

// ---------------------------------------------------------------------------
// Host-side orchestration
// ---------------------------------------------------------------------------
extern "C" void kernel_launch(void* const* d_in, const int* in_sizes, int n_in,
                              void* d_out, int out_size, void* d_ws, size_t ws_size,
                              hipStream_t stream) {
  const float* x     = (const float*)d_in[0];
  const float* W_in  = (const float*)d_in[1];
  const float* b_in  = (const float*)d_in[2];
  const float* A_log = (const float*)d_in[3];
  const float* Dv    = (const float*)d_in[4];
  const float* W_x   = (const float*)d_in[5];
  const float* W_dt  = (const float*)d_in[6];
  const float* b_dt  = (const float*)d_in[7];
  const float* W_out = (const float*)d_in[8];
  const float* b_out = (const float*)d_in[9];
  const float* ln_g  = (const float*)d_in[10];
  const float* ln_b  = (const float*)d_in[11];
  float* out = (float*)d_out;

  char* ws = (char*)d_ws;
  size_t off = 0;
  auto alloc = [&](size_t bytes) -> void* {
    void* p = ws + off;
    off = (off + bytes + 255) & ~(size_t)255;
    return p;
  };

  __bf16* xb     = (__bf16*)alloc((size_t)NT * D_MODEL * 2);
  __bf16* WinT   = (__bf16*)alloc((size_t)D_MODEL * 2 * D_INNER * 2);  // [4096][1024]
  __bf16* WdtT   = (__bf16*)alloc((size_t)DT_RANK * D_INNER * 2);      // [2048][64]
  __bf16* WoutT  = (__bf16*)alloc((size_t)D_INNER * D_MODEL * 2);      // [1024][2048]
  float*  x_in   = (float*)alloc((size_t)NT * D_INNER * 4);
  float*  zbuf   = (float*)alloc((size_t)NT * D_INNER * 4);
  float*  dtbuf  = (float*)alloc((size_t)NT * D_INNER * 4);
  float*  ybuf   = (float*)alloc((size_t)NT * D_INNER * 4);
  __bf16* dtr_b  = (__bf16*)alloc((size_t)NT * DT_RANK * 2);
  float*  Bm     = (float*)alloc((size_t)NT * STATE_DIM * 4);
  float*  Cm     = (float*)alloc((size_t)NT * STATE_DIM * 4);
  __bf16* yb     = (__bf16*)alloc((size_t)NT * D_INNER * 2);

  // 1. operand conversions: x -> bf16; weights -> transposed bf16 [N][K]
  {
    int n = NT * D_MODEL;
    cvt_bf16_kernel<<<(n + 255) / 256, 256, 0, stream>>>(x, xb, n);
  }
  cvt_transpose_kernel<<<dim3(D_MODEL / 32, (2 * D_INNER) / 32), 256, 0, stream>>>(
      W_in, WinT, D_MODEL, 2 * D_INNER);
  cvt_transpose_kernel<<<dim3(DT_RANK / 32, D_INNER / 32), 256, 0, stream>>>(
      W_dt, WdtT, DT_RANK, D_INNER);
  cvt_transpose_kernel<<<dim3(D_INNER / 32, D_MODEL / 32), 256, 0, stream>>>(
      W_out, WoutT, D_INNER, D_MODEL);

  // 2. GEMM1: xz = x @ W_in + b_in -> split into x_in (f32) and z (f32)
  wmma_gemm_kernel<EPI_SPLIT><<<dim3(NT / 64, (2 * D_INNER) / 128), 256, 0, stream>>>(
      xb, WinT, b_in, x_in, zbuf, NT, 2 * D_INNER, D_MODEL, D_MODEL, D_MODEL);

  // 3. x_dbl = x_in @ W_x -> dt_r(bf16), B(f32), C(f32)
  xdbl_kernel<<<NT, 128, 0, stream>>>(x_in, W_x, dtr_b, Bm, Cm);

  // 4. dt = softplus(dt_r @ W_dt + b_dt)
  wmma_gemm_kernel<EPI_SOFTPLUS><<<dim3(NT / 64, D_INNER / 128), 256, 0, stream>>>(
      dtr_b, WdtT, b_dt, dtbuf, nullptr, NT, D_INNER, DT_RANK, DT_RANK, DT_RANK);

  // 5. selective scan
  scan_kernel<<<BATCH * (D_INNER / 256), 256, 0, stream>>>(
      x_in, dtbuf, Bm, Cm, A_log, Dv, ybuf);

  // 6. layernorm + silu gate -> bf16
  ln_gate_kernel<<<NT, 256, 0, stream>>>(ybuf, zbuf, ln_g, ln_b, yb);

  // 7. out = y_gated @ W_out + b_out
  wmma_gemm_kernel<EPI_BIAS><<<dim3(NT / 64, D_MODEL / 128), 256, 0, stream>>>(
      yb, WoutT, b_out, out, nullptr, NT, D_MODEL, D_INNER, D_INNER, D_INNER);
}
